// ViLLayer_75265006895886
// MI455X (gfx1250) — compile-verified, matrix operand order checked by reading, not run on hardware
//
#include <hip/hip_runtime.h>
#include <hip/hip_bf16.h>
#include <math.h>

// ---------------------------------------------------------------------------
// Types / constants
// ---------------------------------------------------------------------------
typedef __bf16 bf16x16 __attribute__((ext_vector_type(16)));
typedef __bf16 bf16x8  __attribute__((ext_vector_type(8)));
typedef __bf16 bf16x4  __attribute__((ext_vector_type(4)));
typedef float  f32x8   __attribute__((ext_vector_type(8)));
typedef float  f32x4   __attribute__((ext_vector_type(4)));

#define WMMA_BF16(A_, B_, C_) \
  __builtin_amdgcn_wmma_f32_16x16x32_bf16(false, (A_), false, (B_), (short)0, (C_), false, false)

constexpr int DIM   = 512;
constexpr int INNER = 1024;
constexpr int NH    = 16;
constexpr int DH    = 64;
constexpr int BB    = 2;
constexpr int SS    = 2048;
constexpr int MROWS = BB * SS;          // 4096
constexpr float EPS      = 5e-5f;
constexpr float NORM_EPS = 1e-6f;
constexpr float CAPV     = 15.0f;

__device__ __forceinline__ float siluf(float x) { return x / (1.0f + __expf(-x)); }

// CDNA5 async global->LDS copy (ASYNCcnt-tracked, no VGPR round trip).
__device__ __forceinline__ void async_lds_b128(unsigned lds_off, const void* gptr) {
  asm volatile("global_load_async_to_lds_b128 %0, %1, off"
               :: "v"(lds_off), "v"(gptr) : "memory");
}
__device__ __forceinline__ void wait_async0() {
  asm volatile("s_wait_asynccnt 0x0" ::: "memory");
}

// ---------------------------------------------------------------------------
// Fragment gather (wave32 WMMA 16-bit layout) from a (rows x K) row-major LDS
// tile: lane row = lane&15, k(e) = 16*(e/8) + 8*(lane>>4) + (e%8).
// Used for A tiles and for B when the B tile is stored (N x K) row-major.
// ---------------------------------------------------------------------------
__device__ __forceinline__ bf16x16 frag_nk(const __bf16* p, int ld, int lx, int half) {
  bf16x16 f;
  const int kb = half * 8;
#pragma unroll
  for (int hh = 0; hh < 2; ++hh)
#pragma unroll
    for (int e = 0; e < 8; ++e)
      f[hh * 8 + e] = p[lx * ld + hh * 16 + kb + e];
  return f;
}

// ---------------------------------------------------------------------------
// Generic WMMA GEMM:  C[M,N] = A[M,K] * W[N,K]^T   (fp32 in, bf16 WMMA path)
// 64x64 block tile, BK=64, 128 threads (4 waves), each wave 32x32.
// Optional bf16 secondary output (used to feed the mLSTM without converts).
// ---------------------------------------------------------------------------
__global__ __launch_bounds__(128) void k_gemm(
    const float* __restrict__ A, int lda, int strideA,
    const float* __restrict__ W, int ldw, int strideW,
    float* __restrict__ C, int ldc, int strideC,
    __bf16* __restrict__ Cb, int ldcb, int strideCb, int K) {
  A += (size_t)blockIdx.z * strideA;
  W += (size_t)blockIdx.z * strideW;
  C += (size_t)blockIdx.z * strideC;
  const int tid  = threadIdx.x;
  const int lane = tid & 31;
  const int w    = tid >> 5;
  const int lx   = lane & 15;
  const int half = lane >> 4;
  const int m0   = blockIdx.x * 64;
  const int n0   = blockIdx.y * 64;
  const int rm   = (w >> 1) * 32;
  const int cn   = (w & 1) * 32;

  __shared__ __bf16 As[64][68];
  __shared__ __bf16 Ws[64][68];

  f32x8 acc[2][2] = {};

  const int rldr = tid >> 4;        // 0..7
  const int rldc = (tid & 15) * 4;  // 0..60

  for (int k0 = 0; k0 < K; k0 += 64) {
    f32x4 ra[8], rw[8];
#pragma unroll
    for (int p = 0; p < 8; ++p) {
      const int r = p * 8 + rldr;
      ra[p] = *(const f32x4*)(A + (size_t)(m0 + r) * lda + k0 + rldc);
      rw[p] = *(const f32x4*)(W + (size_t)(n0 + r) * ldw + k0 + rldc);
    }
    if (k0 + 64 < K)
      __builtin_prefetch(A + (size_t)(m0 + (tid & 63)) * lda + k0 + 64, 0, 1);
    __syncthreads();
#pragma unroll
    for (int p = 0; p < 8; ++p) {
      const int r = p * 8 + rldr;
      *(bf16x4*)&As[r][rldc] = __builtin_convertvector(ra[p], bf16x4);
      *(bf16x4*)&Ws[r][rldc] = __builtin_convertvector(rw[p], bf16x4);
    }
    __syncthreads();
#pragma unroll
    for (int kc = 0; kc < 64; kc += 32) {
      bf16x16 a0 = frag_nk(&As[rm][kc],      68, lx, half);
      bf16x16 a1 = frag_nk(&As[rm + 16][kc], 68, lx, half);
      bf16x16 b0 = frag_nk(&Ws[cn][kc],      68, lx, half);
      bf16x16 b1 = frag_nk(&Ws[cn + 16][kc], 68, lx, half);
      acc[0][0] = WMMA_BF16(a0, b0, acc[0][0]);
      acc[0][1] = WMMA_BF16(a0, b1, acc[0][1]);
      acc[1][0] = WMMA_BF16(a1, b0, acc[1][0]);
      acc[1][1] = WMMA_BF16(a1, b1, acc[1][1]);
    }
  }

#pragma unroll
  for (int i = 0; i < 2; ++i)
#pragma unroll
    for (int j = 0; j < 2; ++j)
#pragma unroll
      for (int r = 0; r < 8; ++r)
        C[(size_t)(m0 + rm + i * 16 + half * 8 + r) * ldc + n0 + cn + j * 16 + lx] =
            acc[i][j][r];
  if (Cb) {
    Cb += (size_t)blockIdx.z * strideCb;
#pragma unroll
    for (int i = 0; i < 2; ++i)
#pragma unroll
      for (int j = 0; j < 2; ++j)
#pragma unroll
        for (int r = 0; r < 8; ++r)
          Cb[(size_t)(m0 + rm + i * 16 + half * 8 + r) * ldcb + n0 + cn + j * 16 + lx] =
              (__bf16)acc[i][j][r];
  }
}

// ---------------------------------------------------------------------------
// Causal depthwise conv1d (KSZ=4) + SiLU.  x_mlstm = x_inner[:, :INNER]
// ---------------------------------------------------------------------------
__global__ void k_conv_silu(const float* __restrict__ x_inner,
                            const float* __restrict__ conv_w,
                            const float* __restrict__ conv_b,
                            float* __restrict__ xca) {
  const int idx = blockIdx.x * blockDim.x + threadIdx.x;
  if (idx >= MROWS * INNER) return;
  const int m = idx >> 10;
  const int c = idx & 1023;
  const int b = m >> 11;
  const int s = m & 2047;
  float y = conv_b[c];
#pragma unroll
  for (int j = 0; j < 4; ++j) {
    const int sp = s - 3 + j;
    if (sp >= 0)
      y += x_inner[((size_t)(b * SS + sp)) * (2 * INNER) + c] * conv_w[c * 4 + j];
  }
  xca[(size_t)m * INNER + c] = siluf(y);
}

// ---------------------------------------------------------------------------
// Gate pre-activations: if_pre = [q,k,v] @ W_if^T + b_if, soft-capped.
// ---------------------------------------------------------------------------
__global__ void k_ifgate(const float* __restrict__ qb, const float* __restrict__ kb,
                         const float* __restrict__ vb, const float* __restrict__ W_if,
                         const float* __restrict__ b_if,
                         float* __restrict__ ig, float* __restrict__ Fc) {
  const int idx = blockIdx.x * blockDim.x + threadIdx.x;
  if (idx >= MROWS * 2 * NH) return;
  const int m = idx >> 5;
  const int g = idx & 31;
  const f32x4* wq = (const f32x4*)(W_if + (size_t)g * (3 * INNER));
  const f32x4* wk = wq + INNER / 4;
  const f32x4* wv = wk + INNER / 4;
  const f32x4* qr = (const f32x4*)(qb + (size_t)m * INNER);
  const f32x4* kr = (const f32x4*)(kb + (size_t)m * INNER);
  const f32x4* vr = (const f32x4*)(vb + (size_t)m * INNER);
  float acc = b_if[g];
  for (int e = 0; e < INNER / 4; ++e) {
    const f32x4 a = qr[e], b4 = wq[e];
    const f32x4 c = kr[e], d4 = wk[e];
    const f32x4 f = vr[e], g4 = wv[e];
    acc += a[0]*b4[0] + a[1]*b4[1] + a[2]*b4[2] + a[3]*b4[3];
    acc += c[0]*d4[0] + c[1]*d4[1] + c[2]*d4[2] + c[3]*d4[3];
    acc += f[0]*g4[0] + f[1]*g4[1] + f[2]*g4[2] + f[3]*g4[3];
  }
  acc = CAPV * tanhf(acc / CAPV);
  const int b = m >> 11;
  const int s = m & 2047;
  if (g < NH)
    ig[((size_t)(b * NH + g)) * SS + s] = acc;
  else
    Fc[((size_t)(b * NH + (g - NH))) * SS + s] = acc;
}

// ---------------------------------------------------------------------------
// In-place: Fc[b,n,:] = cumsum(log_sigmoid(f))
// ---------------------------------------------------------------------------
__global__ void k_cumsum(float* __restrict__ Fc) {
  const int r = blockIdx.x * blockDim.x + threadIdx.x;
  if (r >= BB * NH) return;
  float* p = Fc + (size_t)r * SS;
  float acc = 0.0f;
  for (int s = 0; s < SS; ++s) {
    const float x  = p[s];
    const float ls = (x >= 0.0f) ? -log1pf(__expf(-x)) : (x - log1pf(__expf(x)));
    acc += ls;
    p[s] = acc;
  }
}

// ---------------------------------------------------------------------------
// Flash-style mLSTM parallel kernel. Grid: (S/64, NH, B), 128 threads (4 waves).
// Q/K tiles staged via async global->LDS (bf16, pure copy); V staged transposed
// so both WMMA operand gathers are contiguous LDS reads.
// ---------------------------------------------------------------------------
__global__ __launch_bounds__(128) void k_mlstm(
    const __bf16* __restrict__ qbf, const __bf16* __restrict__ kbf,
    const __bf16* __restrict__ vbf, const float* __restrict__ Fc,
    const float* __restrict__ ig, float* __restrict__ h) {
  const int qt   = blockIdx.x;
  const int head = blockIdx.y;
  const int b    = blockIdx.z;
  const int tid  = threadIdx.x;
  const int lane = tid & 31;
  const int w    = tid >> 5;
  const int lx   = lane & 15;
  const int half = lane >> 4;

  __shared__ __bf16 Qs[64][64];
  __shared__ __bf16 Ks[64][64];
  __shared__ __bf16 Vt[64][72];   // transposed: Vt[d][t]
  __shared__ __bf16 Cs[4][16][64];
  __shared__ float  Fk[64];
  __shared__ float  Ik[64];

  const int  qbase = qt * 64;
  const long mrow0 = (long)b * SS + qbase;
  const long gbase = ((long)b * NH + head) * SS;
  const unsigned qs_base = (unsigned)(size_t)&Qs[0][0];
  const unsigned ks_base = (unsigned)(size_t)&Ks[0][0];

  // Async-stage Q tile (unscaled bf16; DH^-0.5 folded into C below).
#pragma unroll
  for (int p = 0; p < 4; ++p) {
    const int u = p * 128 + tid;          // 16B unit
    const int r = u >> 3, c8 = (u & 7) * 8;
    async_lds_b128(qs_base + (unsigned)u * 16,
                   qbf + (mrow0 + r) * INNER + head * DH + c8);
  }

  float Fq[8];
#pragma unroll
  for (int r = 0; r < 8; ++r)
    Fq[r] = Fc[gbase + qbase + w * 16 + half * 8 + r];

  f32x8 o[4] = {};
  float m_[8], sC[8];
#pragma unroll
  for (int r = 0; r < 8; ++r) { m_[r] = -INFINITY; sC[r] = 0.0f; }

  for (int kt = 0; kt <= qt; ++kt) {
    const int kb = kt * 64;
    __syncthreads();   // previous tile fully consumed by all waves
#pragma unroll
    for (int p = 0; p < 4; ++p) {
      const int u = p * 128 + tid;
      const int r = u >> 3, c8 = (u & 7) * 8;
      async_lds_b128(ks_base + (unsigned)u * 16,
                     kbf + ((long)b * SS + kb + r) * INNER + head * DH + c8);
    }
#pragma unroll
    for (int p = 0; p < 4; ++p) {
      const int u  = p * 128 + tid;
      const int t  = u >> 3, d0 = (u & 7) * 8;
      bf16x8 vv = *(const bf16x8*)(vbf + ((long)b * SS + kb + t) * INNER + head * DH + d0);
#pragma unroll
      for (int j = 0; j < 8; ++j) Vt[d0 + j][t] = vv[j];
    }
    if (tid < 64) {
      Fk[tid] = Fc[gbase + kb + tid];
      Ik[tid] = ig[gbase + kb + tid];
    }
    wait_async0();
    __syncthreads();

    // ---- S = Q K^T ----
    f32x8 sacc[4] = {};
#pragma unroll
    for (int kc = 0; kc < 64; kc += 32) {
      bf16x16 a = frag_nk(&Qs[w * 16][kc], 64, lx, half);
#pragma unroll
      for (int j = 0; j < 4; ++j) {
        bf16x16 bf = frag_nk(&Ks[j * 16][kc], 64, lx, half);
        sacc[j] = WMMA_BF16(a, bf, sacc[j]);
      }
    }

    // ---- gating ----
    float logd[4][8];
    float rmax[8];
#pragma unroll
    for (int r = 0; r < 8; ++r) rmax[r] = -INFINITY;
#pragma unroll
    for (int j = 0; j < 4; ++j) {
      const int   tcol = kb + j * 16 + lx;
      const float fkt  = Fk[j * 16 + lx];
      const float ikt  = Ik[j * 16 + lx];
#pragma unroll
      for (int r = 0; r < 8; ++r) {
        const int srow = qbase + w * 16 + half * 8 + r;
        const float v  = (tcol <= srow) ? (Fq[r] - fkt + ikt) : -INFINITY;
        logd[j][r] = v;
        rmax[r]    = fmaxf(rmax[r], v);
      }
    }
#pragma unroll
    for (int r = 0; r < 8; ++r) {
      rmax[r] = fmaxf(rmax[r], __shfl_xor(rmax[r], 1, 32));
      rmax[r] = fmaxf(rmax[r], __shfl_xor(rmax[r], 2, 32));
      rmax[r] = fmaxf(rmax[r], __shfl_xor(rmax[r], 4, 32));
      rmax[r] = fmaxf(rmax[r], __shfl_xor(rmax[r], 8, 32));
    }
    float scl[8];
#pragma unroll
    for (int r = 0; r < 8; ++r) {
      const float mn = fmaxf(m_[r], rmax[r]);
      scl[r] = (m_[r] == -INFINITY) ? 0.0f : __expf(m_[r] - mn);
      m_[r]  = mn;
    }

    // ---- C = (S * DH^-0.5) * exp(logD - m); row-sum; spill to LDS ----
    float rsum[8];
#pragma unroll
    for (int r = 0; r < 8; ++r) rsum[r] = 0.0f;
#pragma unroll
    for (int j = 0; j < 4; ++j)
#pragma unroll
      for (int r = 0; r < 8; ++r) {
        const float c = (0.125f * sacc[j][r]) * __expf(logd[j][r] - m_[r]);
        rsum[r] += c;
        Cs[w][half * 8 + r][j * 16 + lx] = (__bf16)c;
      }
#pragma unroll
    for (int r = 0; r < 8; ++r) {
      rsum[r] += __shfl_xor(rsum[r], 1, 32);
      rsum[r] += __shfl_xor(rsum[r], 2, 32);
      rsum[r] += __shfl_xor(rsum[r], 4, 32);
      rsum[r] += __shfl_xor(rsum[r], 8, 32);
      sC[r] = sC[r] * scl[r] + rsum[r];
    }
#pragma unroll
    for (int j = 0; j < 4; ++j)
#pragma unroll
      for (int r = 0; r < 8; ++r)
        o[j][r] = o[j][r] * scl[r];

    // ---- O += C @ V (B gather from transposed Vt rows: contiguous) ----
#pragma unroll
    for (int kc = 0; kc < 64; kc += 32) {
      bf16x16 a = frag_nk(&Cs[w][0][kc], 64, lx, half);
#pragma unroll
      for (int j = 0; j < 4; ++j) {
        bf16x16 bf = frag_nk(&Vt[j * 16][kc], 72, lx, half);
        o[j] = WMMA_BF16(a, bf, o[j]);
      }
    }
  }

  // ---- finalize: h = O / (max(|sumC|, exp(-m)) + eps) ----
  float inv[8];
#pragma unroll
  for (int r = 0; r < 8; ++r) {
    const float n = fmaxf(fabsf(sC[r]), __expf(-m_[r]));
    inv[r] = 1.0f / (n + EPS);
  }
#pragma unroll
  for (int j = 0; j < 4; ++j)
#pragma unroll
    for (int r = 0; r < 8; ++r) {
      const int srow = qbase + w * 16 + half * 8 + r;
      h[(gbase + srow) * DH + j * 16 + lx] = o[j][r] * inv[r];
    }
}

// ---------------------------------------------------------------------------
// RMS-norm per head, gain, skip, SiLU(z) gate. One thread per (b,s,n) row.
// ---------------------------------------------------------------------------
__global__ void k_norm_merge(const float* __restrict__ h,
                             const float* __restrict__ x_inner,
                             const float* __restrict__ xca,
                             const float* __restrict__ norm_w,
                             const float* __restrict__ skip,
                             float* __restrict__ hout) {
  const int rid = blockIdx.x * blockDim.x + threadIdx.x;
  if (rid >= BB * NH * SS) return;
  const int bn = rid / SS;
  const int s  = rid % SS;
  const int n  = bn % NH;
  const int b  = bn / NH;
  const float* hr = h + (size_t)rid * DH;
  float ss = 0.0f;
#pragma unroll 8
  for (int d = 0; d < DH; ++d) ss += hr[d] * hr[d];
  const float rrms = rsqrtf(ss / (float)DH + NORM_EPS);
  const size_t m = (size_t)b * SS + s;
#pragma unroll 8
  for (int d = 0; d < DH; ++d) {
    const int   c  = n * DH + d;
    const float hn = hr[d] * rrms * (1.0f + norm_w[c]);
    const float ht = hn + skip[c] * xca[m * INNER + c];
    const float z  = x_inner[m * (2 * INNER) + INNER + c];
    hout[m * INNER + c] = ht * siluf(z);
  }
}

// ---------------------------------------------------------------------------
// Launch
// ---------------------------------------------------------------------------
extern "C" void kernel_launch(void* const* d_in, const int* in_sizes, int n_in,
                              void* d_out, int out_size, void* d_ws, size_t ws_size,
                              hipStream_t stream) {
  (void)in_sizes; (void)n_in; (void)out_size; (void)ws_size;
  const float* x      = (const float*)d_in[0];
  const float* W_up   = (const float*)d_in[1];
  const float* W_q    = (const float*)d_in[2];
  const float* W_k    = (const float*)d_in[3];
  const float* W_v    = (const float*)d_in[4];
  const float* conv_w = (const float*)d_in[5];
  const float* conv_b = (const float*)d_in[6];
  const float* W_if   = (const float*)d_in[7];
  const float* b_if   = (const float*)d_in[8];
  const float* norm_w = (const float*)d_in[9];
  const float* skip   = (const float*)d_in[10];
  const float* W_down = (const float*)d_in[11];
  float* out = (float*)d_out;

  float* ws      = (float*)d_ws;
  float* x_inner = ws;  ws += (size_t)MROWS * 2 * INNER;
  float* xca     = ws;  ws += (size_t)MROWS * INNER;
  float* qb      = ws;  ws += (size_t)MROWS * INNER;
  float* kb      = ws;  ws += (size_t)MROWS * INNER;
  float* vb      = ws;  ws += (size_t)MROWS * INNER;
  float* hb      = ws;  ws += (size_t)MROWS * INNER;
  float* hout    = ws;  ws += (size_t)MROWS * INNER;
  float* ig      = ws;  ws += (size_t)BB * NH * SS;
  float* Fc      = ws;  ws += (size_t)BB * NH * SS;
  __bf16* qbf    = (__bf16*)ws;
  __bf16* kbf    = qbf + (size_t)MROWS * INNER;
  __bf16* vbf    = kbf + (size_t)MROWS * INNER;

  const dim3 blk(128);

  // 1. up projection: x_inner = x @ W_up^T  (4096 x 2048)
  k_gemm<<<dim3(MROWS / 64, (2 * INNER) / 64, 1), blk, 0, stream>>>(
      x, DIM, 0, W_up, DIM, 0, x_inner, 2 * INNER, 0, nullptr, 0, 0, DIM);

  // 2. causal conv + SiLU
  k_conv_silu<<<(MROWS * INNER) / 256, 256, 0, stream>>>(x_inner, conv_w, conv_b, xca);

  // 3-5. head-wise q/k/v projections (fp32 + bf16 outputs)
  k_gemm<<<dim3(MROWS / 64, 1, NH), blk, 0, stream>>>(
      xca, INNER, DH, W_q, DH, DH * DH, qb, INNER, DH, qbf, INNER, DH, DH);
  k_gemm<<<dim3(MROWS / 64, 1, NH), blk, 0, stream>>>(
      xca, INNER, DH, W_k, DH, DH * DH, kb, INNER, DH, kbf, INNER, DH, DH);
  k_gemm<<<dim3(MROWS / 64, 1, NH), blk, 0, stream>>>(
      x_inner, 2 * INNER, DH, W_v, DH, DH * DH, vb, INNER, DH, vbf, INNER, DH, DH);

  // 6. gates (capped) + log-sigmoid cumsum
  k_ifgate<<<(MROWS * 2 * NH) / 256, 256, 0, stream>>>(qb, kb, vb, W_if, b_if, ig, Fc);
  k_cumsum<<<1, 32, 0, stream>>>(Fc);

  // 7. mLSTM attention core (bf16 operands, fp32 gating/normalizer)
  k_mlstm<<<dim3(SS / 64, NH, BB), blk, 0, stream>>>(qbf, kbf, vbf, Fc, ig, hb);

  // 8. norm + skip + SiLU(z) merge
  k_norm_merge<<<(MROWS * NH) / 256, 256, 0, stream>>>(hb, x_inner, xca, norm_w, skip, hout);

  // 9. down projection -> output
  k_gemm<<<dim3(MROWS / 64, DIM / 64, 1), blk, 0, stream>>>(
      hout, INNER, 0, W_down, INNER, 0, out, DIM, 0, nullptr, 0, 0, INNER);
}
